// TRANS_BLOCK_4750233829697
// MI455X (gfx1250) — matrix operand order repaired and checked by reading, not compiled, not run on hardware
//
#include <hip/hip_runtime.h>
#include <hip/hip_bf16.h>
#include <math.h>

// ---------------------------------------------------------------------------
// CDNA5 (gfx1250) transformer block: attention + conv(1,9)+BN+res+relu + FF.
// All GEMMs via v_wmma_f32_16x16x32_f16 (wave32, 16x16 tiles).
// Attention stages K/V tiles into LDS with global_load_async_to_lds_b128
// (ASYNCcnt) shared by 4 waves per workgroup, double-buffered.
// ---------------------------------------------------------------------------

typedef __attribute__((ext_vector_type(16))) _Float16 v16h;
typedef __attribute__((ext_vector_type(8)))  _Float16 half8;
typedef __attribute__((ext_vector_type(8)))  float    v8f;

#define EPSV 1e-5f
#define NB 16
#define CH 64
#define TT 64
#define VVV 25
#define LL 1600   // TT*VVV
#define NH 4
#define DD 64
#define QKVJ 768  // 3*NH*DD

// A-fragment (16x32 f16, M = lane%16): lane's 16 halfs are a contiguous run
// K = kbase + 16*(lane/16) + {0..15}, matching the B-fragment order
__device__ inline v16h load_a_frag(const _Float16* rowBase, int kbase, int lane) {
  int klo = kbase + ((lane & 16) ? 16 : 0);
  half8 lo = *reinterpret_cast<const half8*>(rowBase + klo);
  half8 hi = *reinterpret_cast<const half8*>(rowBase + klo + 8);
  v16h a;
#pragma unroll
  for (int i = 0; i < 8; ++i) { a[i] = lo[i]; a[i + 8] = hi[i]; }
  return a;
}

// B-fragment (32x16 f16, N = lane%16): lane holds column N of B, i.e. a
// contiguous run of 16 halfs of B^T row N: K = kbase + 16*(lane/16) + {0..15}
__device__ inline v16h load_b_frag(const _Float16* rowBase, int kbase, int lane) {
  return *reinterpret_cast<const v16h*>(rowBase + kbase + ((lane & 16) ? 16 : 0));
}

__device__ inline v8f wmma_f16(v16h a, v16h b, v8f c) {
  return __builtin_amdgcn_wmma_f32_16x16x32_f16(false, a, false, b, (short)0, c,
                                                false, false);
}

// Async copy 16B per lane from global to LDS (CDNA5 ASYNCcnt path).
__device__ inline void async_g2l_b128(unsigned ldsOff, const void* gsrc) {
  asm volatile("global_load_async_to_lds_b128 %0, %1, off"
               :: "v"(ldsOff), "v"(gsrc) : "memory");
}
__device__ inline unsigned lds_off(const void* p) {
  // LDS aperture: low 32 bits of a flat shared pointer are the LDS byte offset
  return (unsigned)(unsigned long long)p;
}

// ------------------------- prep: weights / BN fold -------------------------
__global__ void tb_prep(const float* Wqkv, const float* Wout, const float* Wff,
                        const float* g1, const float* be1, const float* mu1,
                        const float* va1, const float* g2, const float* be2,
                        const float* mu2, const float* va2,
                        _Float16* WtQ, _Float16* Wdv, _Float16* Wff16, float* bn) {
  const int n0 = QKVJ * CH;          // W_qkv^T  [768][64]
  const int n1 = 9 * CH * 256;       // W_out -> [dv][co][ci]
  const int n2 = CH * CH;            // W_ff     [64][64]
  const int total = n0 + n1 + n2 + 2 * CH;
  for (int i = blockIdx.x * blockDim.x + threadIdx.x; i < total;
       i += gridDim.x * blockDim.x) {
    int t = i;
    if (t < n0) { int j = t / CH, c = t % CH;
      WtQ[t] = (_Float16)Wqkv[c * QKVJ + j]; continue; }
    t -= n0;
    if (t < n1) { int dv = t / (CH * 256); int r = t % (CH * 256);
      int co = r / 256, ci = r % 256;
      Wdv[t] = (_Float16)Wout[(co * 256 + ci) * 9 + dv]; continue; }
    t -= n1;
    if (t < n2) { Wff16[t] = (_Float16)Wff[t]; continue; }
    t -= n2;
    if (t < CH) { float s = g1[t] * rsqrtf(va1[t] + EPSV);
      bn[t] = s; bn[64 + t] = be1[t] - mu1[t] * s; }
    else { int c = t - CH; float s = g2[c] * rsqrtf(va2[c] + EPSV);
      bn[128 + c] = s; bn[192 + c] = be2[c] - mu2[c] * s; }
  }
}

// ------------------------- prep: x -> x^T (f16) ----------------------------
__global__ void tb_xt(const float* x, _Float16* xT) {
  int i = blockIdx.x * blockDim.x + threadIdx.x;
  if (i >= NB * CH * LL) return;
  int n = i / (CH * LL); int r = i % (CH * LL);
  int c = r / LL; int l = r % LL;
  xT[((size_t)n * LL + l) * CH + c] = (_Float16)x[i];
}

// ---------------- QKV projection: [1600x64] @ [64x768] per batch -----------
__global__ __launch_bounds__(32)
void tb_qkv(const _Float16* xT, const _Float16* WtQ, const float* bqkv,
            _Float16* q16, _Float16* k16, _Float16* vT) {
  int lane = threadIdx.x;
  int bid = blockIdx.x;
  int nt = bid % 48; int tmp = bid / 48;
  int mt = tmp % 100; int n = tmp / 100;
  const _Float16* arow = xT + ((size_t)n * LL + mt * 16 + (lane & 15)) * CH;
  const _Float16* brow = WtQ + (size_t)(nt * 16 + (lane & 15)) * CH;
  v8f c = {};
#pragma unroll
  for (int s = 0; s < 2; ++s)
    c = wmma_f16(load_a_frag(arow, s * 32, lane),
                 load_b_frag(brow, s * 32, lane), c);
  int j = nt * 16 + (lane & 15);
  int which = j >> 8; int h = (j >> 6) & 3; int d = j & 63;
  float bias = bqkv[j];
  int hi8 = (lane & 16) ? 8 : 0;
#pragma unroll
  for (int r = 0; r < 8; ++r) {
    int l = mt * 16 + r + hi8;
    _Float16 hv = (_Float16)(c[r] + bias);
    if (which == 0)      q16[((size_t)(n * NH + h) * LL + l) * DD + d] = hv;
    else if (which == 1) k16[((size_t)(n * NH + h) * LL + l) * DD + d] = hv;
    else                 vT[((size_t)(n * NH + h) * DD + d) * LL + l] = hv;
  }
}

// ---------------- Flash attention: 4 waves/block, LDS-staged K/V -----------
// Block = 128 threads; wave w owns query tile mt_base+w. Per 32-key step the
// K tile [32x64] and V^T tile [64x32] are async-copied into LDS (double
// buffered) and consumed by all 4 waves.
__global__ __launch_bounds__(128)
void tb_attn(const _Float16* q16, const _Float16* k16, const _Float16* vT,
             _Float16* oT) {
  __shared__ __align__(32) _Float16 kbuf[2][32 * 64];  // [key][d]   4KB each
  __shared__ __align__(32) _Float16 vbuf[2][64 * 32];  // [d][key]   4KB each
  __shared__ __align__(32) _Float16 pbuf[4][16 * 32];  // per-wave P tile
  int tid = threadIdx.x;
  int lane = tid & 31;
  int w = tid >> 5;
  int bid = blockIdx.x;
  int mt = (bid % 25) * 4 + w;
  int tmp = bid / 25; int h = tmp & 3; int n = tmp >> 2;
  size_t headoff = (size_t)(n * NH + h) * LL * DD;

  const _Float16* qrow = q16 + headoff + (size_t)(mt * 16 + (lane & 15)) * DD;
  v16h aq0 = load_a_frag(qrow, 0, lane);
  v16h aq1 = load_a_frag(qrow, 32, lane);
  v8f acc[4];
#pragma unroll
  for (int t = 0; t < 4; ++t) acc[t] = {};
  float mrow[8], lrow[8];
#pragma unroll
  for (int r = 0; r < 8; ++r) { mrow[r] = -1e30f; lrow[r] = 0.f; }
  const float scale = 0.125f;   // 1/sqrt(64)
  int hi8 = (lane & 16) ? 8 : 0;
  int vrow_id = tid >> 2, vsub = tid & 3;  // for V staging

  for (int kb = 0; kb < 50; ++kb) {          // 32 keys per step
    int buf = kb & 1;
    // --- stage K tile (contiguous 4KB) and V^T tile (64 rows x 64B) -------
    {
      unsigned kd = lds_off(&kbuf[buf][0]) + tid * 16;
      const char* kg = (const char*)(k16 + headoff + (size_t)kb * 32 * DD)
                       + tid * 16;
      async_g2l_b128(kd, kg);
      async_g2l_b128(kd + 2048u, kg + 2048);
      unsigned vd = lds_off(&vbuf[buf][0]) + tid * 16;
      const char* vg = (const char*)(vT + headoff + (size_t)kb * 32)
                       + (size_t)vrow_id * (LL * 2) + vsub * 16;
      async_g2l_b128(vd, vg);
      async_g2l_b128(vd + 2048u, vg + (size_t)32 * (LL * 2));
    }
    asm volatile("s_wait_asynccnt 0" ::: "memory");
    __syncthreads();

    // --- S = Q K^T from LDS ----------------------------------------------
    const _Float16* krow0 = &kbuf[buf][(lane & 15) * 64];
    const _Float16* krow1 = krow0 + 16 * 64;
    v8f s0 = {}, s1 = {};
    s0 = wmma_f16(aq0, load_b_frag(krow0, 0, lane), s0);
    s0 = wmma_f16(aq1, load_b_frag(krow0, 32, lane), s0);
    s1 = wmma_f16(aq0, load_b_frag(krow1, 0, lane), s1);
    s1 = wmma_f16(aq1, load_b_frag(krow1, 32, lane), s1);

    // --- online softmax: row M = r + hi8 lives in element r across 16 lanes
#pragma unroll
    for (int r = 0; r < 8; ++r) {
      float x0 = s0[r] * scale, x1 = s1[r] * scale;
      float mx = fmaxf(x0, x1);
#pragma unroll
      for (int msk = 1; msk < 16; msk <<= 1)
        mx = fmaxf(mx, __shfl_xor(mx, msk, 32));
      float newm = fmaxf(mrow[r], mx);
      float p0 = __expf(x0 - newm);
      float p1 = __expf(x1 - newm);
      float rs = p0 + p1;
#pragma unroll
      for (int msk = 1; msk < 16; msk <<= 1) rs += __shfl_xor(rs, msk, 32);
      float corr = __expf(mrow[r] - newm);
      lrow[r] = lrow[r] * corr + rs;
      mrow[r] = newm;
#pragma unroll
      for (int t = 0; t < 4; ++t) acc[t][r] *= corr;
      int M = r + hi8;
      pbuf[w][M * 32 + (lane & 15)]      = (_Float16)p0;
      pbuf[w][M * 32 + 16 + (lane & 15)] = (_Float16)p1;
    }
    asm volatile("s_wait_dscnt 0" ::: "memory");  // wave-local LDS RAW

    v16h pa = load_a_frag(&pbuf[w][(lane & 15) * 32], 0, lane);
#pragma unroll
    for (int t = 0; t < 4; ++t) {
      const _Float16* vrow = &vbuf[buf][(t * 16 + (lane & 15)) * 32];
      acc[t] = wmma_f16(pa, load_b_frag(vrow, 0, lane), acc[t]);
    }
    // double buffering + the barrier above keeps buffers race-free
  }
#pragma unroll
  for (int r = 0; r < 8; ++r) {
    float inv = 1.0f / lrow[r];
    int l = mt * 16 + r + hi8;
#pragma unroll
    for (int t = 0; t < 4; ++t) {
      int d = t * 16 + (lane & 15);
      oT[((size_t)n * LL + l) * 256 + h * DD + d] = (_Float16)(acc[t][r] * inv);
    }
  }
}

// -------- conv(1,9) as 9 shifted GEMMs + bias + BN1 + residual + relu ------
__global__ __launch_bounds__(32)
void tb_conv(const _Float16* Wdv, const _Float16* oT, const float* bout,
             const float* bn, const float* x, float* y32, _Float16* yT) {
  int lane = threadIdx.x;
  int bid = blockIdx.x;
  int lt = bid % 100; int tmp = bid / 100;
  int ct = tmp & 3; int n = tmp >> 2;
  int l = lt * 16 + (lane & 15);
  int t0 = l / VVV, v = l % VVV;
  v8f acc = {};
  const _Float16* arowbase = Wdv + (size_t)(ct * 16 + (lane & 15)) * 256;
#pragma unroll
  for (int dv = 0; dv < 9; ++dv) {
    int v2 = v + dv - 4;
    bool valid = (v2 >= 0) && (v2 < VVV);
    int srcl = t0 * VVV + v2;
    const _Float16* arow = arowbase + (size_t)dv * CH * 256;
    const _Float16* brow = oT + ((size_t)n * LL + srcl) * 256;
    for (int kk = 0; kk < 8; ++kk) {           // K = 256
      v16h a = load_a_frag(arow, kk * 32, lane);
      v16h b = {};
      if (valid) b = load_b_frag(brow, kk * 32, lane);
      acc = wmma_f16(a, b, acc);
    }
  }
  int hi8 = (lane & 16) ? 8 : 0;
#pragma unroll
  for (int r = 0; r < 8; ++r) {
    int co = ct * 16 + r + hi8;
    float s1 = bn[co], h1 = bn[64 + co];
    float val = (acc[r] + bout[co]) * s1 + h1 + x[((size_t)n * CH + co) * LL + l];
    val = fmaxf(val, 0.f);
    y32[((size_t)n * CH + co) * LL + l] = val;
    yT[((size_t)n * LL + l) * CH + co] = (_Float16)val;
  }
}

// -------- 1x1 FF GEMM + bias + BN2 + residual + relu -> output -------------
__global__ __launch_bounds__(32)
void tb_ff(const _Float16* Wff16, const _Float16* yT, const float* bff,
           const float* bn, const float* y32, float* out) {
  int lane = threadIdx.x;
  int bid = blockIdx.x;
  int lt = bid % 100; int tmp = bid / 100;
  int ct = tmp & 3; int n = tmp >> 2;
  int l = lt * 16 + (lane & 15);
  const _Float16* arow = Wff16 + (size_t)(ct * 16 + (lane & 15)) * CH;
  const _Float16* brow = yT + ((size_t)n * LL + l) * CH;
  v8f acc = {};
#pragma unroll
  for (int s = 0; s < 2; ++s)
    acc = wmma_f16(load_a_frag(arow, s * 32, lane),
                   load_b_frag(brow, s * 32, lane), acc);
  int hi8 = (lane & 16) ? 8 : 0;
#pragma unroll
  for (int r = 0; r < 8; ++r) {
    int co = ct * 16 + r + hi8;
    float s2 = bn[128 + co], h2 = bn[192 + co];
    float yv = y32[((size_t)n * CH + co) * LL + l];
    float val = (acc[r] + bff[co]) * s2 + h2 + yv;
    out[((size_t)n * CH + co) * LL + l] = fmaxf(val, 0.f);
  }
}

// ---------------------------------------------------------------------------
extern "C" void kernel_launch(void* const* d_in, const int* in_sizes, int n_in,
                              void* d_out, int out_size, void* d_ws, size_t ws_size,
                              hipStream_t stream) {
  const float* x     = (const float*)d_in[0];
  const float* Wqkv  = (const float*)d_in[1];
  const float* bqkv  = (const float*)d_in[2];
  const float* Wout  = (const float*)d_in[3];
  const float* bout  = (const float*)d_in[4];
  const float* g1    = (const float*)d_in[5];
  const float* be1   = (const float*)d_in[6];
  const float* mu1   = (const float*)d_in[7];
  const float* va1   = (const float*)d_in[8];
  const float* Wff   = (const float*)d_in[9];
  const float* bff   = (const float*)d_in[10];
  const float* g2    = (const float*)d_in[11];
  const float* be2   = (const float*)d_in[12];
  const float* mu2   = (const float*)d_in[13];
  const float* va2   = (const float*)d_in[14];
  float* out = (float*)d_out;

  char* ws = (char*)d_ws;
  size_t off = 0;
  auto alloc = [&](size_t bytes) -> void* {
    void* p = ws + off;
    off += (bytes + 255) & ~(size_t)255;
    return p;
  };
  _Float16* xT    = (_Float16*)alloc((size_t)NB * LL * CH * 2);
  _Float16* WtQ   = (_Float16*)alloc((size_t)QKVJ * CH * 2);
  _Float16* Wdv   = (_Float16*)alloc((size_t)9 * CH * 256 * 2);
  _Float16* Wff16 = (_Float16*)alloc((size_t)CH * CH * 2);
  float*    bn    = (float*)alloc(256 * 4);
  _Float16* q16   = (_Float16*)alloc((size_t)NB * NH * LL * DD * 2);
  _Float16* k16   = (_Float16*)alloc((size_t)NB * NH * LL * DD * 2);
  _Float16* vT    = (_Float16*)alloc((size_t)NB * NH * DD * LL * 2);
  _Float16* oT    = (_Float16*)alloc((size_t)NB * LL * 256 * 2);
  float*    y32   = (float*)alloc((size_t)NB * CH * LL * 4);
  _Float16* yT    = (_Float16*)alloc((size_t)NB * LL * CH * 2);

  tb_prep<<<512, 256, 0, stream>>>(Wqkv, Wout, Wff, g1, be1, mu1, va1,
                                   g2, be2, mu2, va2, WtQ, Wdv, Wff16, bn);
  tb_xt<<<(NB * CH * LL + 255) / 256, 256, 0, stream>>>(x, xT);
  tb_qkv<<<NB * 100 * 48, 32, 0, stream>>>(xT, WtQ, bqkv, q16, k16, vT);
  tb_attn<<<NB * NH * 25, 128, 0, stream>>>(q16, k16, vT, oT);
  tb_conv<<<NB * 4 * 100, 32, 0, stream>>>(Wdv, oT, bout, bn, x, y32, yT);
  tb_ff<<<NB * 4 * 100, 32, 0, stream>>>(Wff16, yT, bff, bn, y32, out);
}
